// looc_layer_54425825575208
// MI455X (gfx1250) — matrix operand order repaired and checked by reading, not compiled, not run on hardware
//
#include <hip/hip_runtime.h>

// ---------------------------------------------------------------------------
// out[b,o] = sqrt(|x_b|^2 - 2 x_b.w_o + |w_o|^2), B=8192 IN=4096 OUT=2048
// GEMM (137 GFLOP) on the bf16 WMMA pipe; norms + epilogue in exact fp32.
// Pipeline (big-ws path): prep kernels convert x->bf16 (+row norms) and
// w->bf16 TRANSPOSED [OUT][IN] (so both GEMM operands stage as [.][K] with
// aligned b128 copies); main kernel double-buffers LDS with async
// global->LDS loads (ASYNCcnt) overlapped with 16 wmma per K-step.
// ---------------------------------------------------------------------------

typedef __attribute__((ext_vector_type(16))) __bf16        v16bf;
typedef __attribute__((ext_vector_type(8)))  float         v8f;
typedef __attribute__((ext_vector_type(4)))  unsigned int  uint4v;
typedef __attribute__((ext_vector_type(4)))  int           int4v;

#define B_DIM   8192
#define IN_DIM  4096
#define OUT_DIM 2048

#define BM 256      // block tile M
#define BN 128      // block tile N
#define BK 32       // K step (one bf16 WMMA K)
#define AST 40      // LDS row stride in bf16 elems (80B -> clean 64-bank spread)
#define BST 40

// ---- async global->LDS path (gfx1250), guarded --------------------------
#if defined(__has_builtin)
#if __has_builtin(__builtin_amdgcn_global_load_async_to_lds_b128) && \
    __has_builtin(__builtin_amdgcn_s_wait_asynccnt)
#define USE_ASYNC 1
#endif
#endif
#ifndef USE_ASYNC
#define USE_ASYNC 0
#endif

#if USE_ASYNC
typedef __attribute__((address_space(1))) int4v glb_int4;   // global V4i
typedef __attribute__((address_space(3))) int4v lds_int4;   // LDS V4i
__device__ __forceinline__ void async_ld16(const unsigned short* g, unsigned short* l) {
    // flat-address low 32 bits == LDS byte address (ISA 10.2 aperture rule)
    __builtin_amdgcn_global_load_async_to_lds_b128(
        (glb_int4*)(size_t)g,
        (lds_int4*)(unsigned)(size_t)l,
        0, 0);
}
#endif

// fp32 -> bf16 RNE via compiler cast (hardware cvt where available)
__device__ __forceinline__ unsigned short bfbits(float f) {
    union { __bf16 h; unsigned short s; } u; u.h = (__bf16)f; return u.s;
}
// manual RNE bit path (used by the fp32-inline fallback GEMM)
__device__ __forceinline__ unsigned short f2bf(float f) {
    unsigned int u = __builtin_bit_cast(unsigned int, f);
    u += 0x7FFFu + ((u >> 16) & 1u);
    return (unsigned short)(u >> 16);
}

union ABFrag { v16bf v; uint4v u[2]; };

// ---- prep: x -> bf16 (same layout) fused with |x_b|^2 --------------------
__global__ void looc_xprep(const float* __restrict__ x, unsigned short* __restrict__ xb,
                           float* __restrict__ x2) {
    const int wave = threadIdx.x >> 5;
    const int lane = threadIdx.x & 31;
    const size_t row = (size_t)blockIdx.x * 8 + wave;
    const float* xr = x + row * IN_DIM;
    unsigned short* xo = xb + row * IN_DIM;
    float s = 0.0f;
    for (int c4 = lane; c4 < IN_DIM / 4; c4 += 32) {
        const float4 v = *(const float4*)(xr + c4 * 4);
        s += v.x * v.x + v.y * v.y + v.z * v.z + v.w * v.w;
        union { unsigned short h[4]; uint2 u; } p;
        p.h[0] = bfbits(v.x); p.h[1] = bfbits(v.y);
        p.h[2] = bfbits(v.z); p.h[3] = bfbits(v.w);
        *(uint2*)(xo + c4 * 4) = p.u;
    }
    #pragma unroll
    for (int off = 16; off > 0; off >>= 1) s += __shfl_xor(s, off, 32);
    if (lane == 0) x2[row] = s;
}

// ---- prep: w [IN][OUT] fp32 -> wT [OUT][IN] bf16, LDS-tiled transpose ----
__global__ void looc_wprep(const float* __restrict__ w, unsigned short* __restrict__ wT) {
    __shared__ unsigned short t[64][68];      // +4 pad: conflict-free transpose
    const int i0 = blockIdx.y * 64;           // IN tile origin
    const int o0 = blockIdx.x * 64;           // OUT tile origin
    const int c  = threadIdx.x & 63;
    const int r0 = threadIdx.x >> 6;          // 0..3
    #pragma unroll
    for (int j = 0; j < 16; ++j) {
        int r = r0 + j * 4;
        float v = w[(size_t)(i0 + r) * OUT_DIM + (o0 + c)];   // coalesced read
        t[c][r] = bfbits(v);                                  // transposed in LDS
    }
    __syncthreads();
    #pragma unroll
    for (int j = 0; j < 16; ++j) {
        int rr = r0 + j * 4;
        wT[(size_t)(o0 + rr) * IN_DIM + (i0 + c)] = t[rr][c]; // coalesced write
    }
}

// ---- |w_o|^2 -------------------------------------------------------------
__global__ void looc_colnorm(const float* __restrict__ w, float* __restrict__ w2) {
    const int c = blockIdx.x * blockDim.x + threadIdx.x;
    float s = 0.0f;
    for (int i = 0; i < IN_DIM; ++i) { float v = w[(size_t)i * OUT_DIM + c]; s += v * v; }
    w2[c] = s;
}

// ---- |x_b|^2 only (small-ws fallback) ------------------------------------
__global__ void looc_rownorm(const float* __restrict__ x, float* __restrict__ x2) {
    const int wave = threadIdx.x >> 5;
    const int lane = threadIdx.x & 31;
    const int row  = blockIdx.x * 8 + wave;
    const float* xr = x + (size_t)row * IN_DIM;
    float s = 0.0f;
    for (int c = lane; c < IN_DIM; c += 32) { float v = xr[c]; s += v * v; }
    #pragma unroll
    for (int off = 16; off > 0; off >>= 1) s += __shfl_xor(s, off, 32);
    if (lane == 0) x2[row] = s;
}

// ===========================================================================
// Main GEMM, bf16 pre-converted inputs: xb [B][IN], wT [OUT][IN].
// Block 256x128, 8 waves (4Mx2N), wave tile 64x64 = 16 wmma / K-step.
// ===========================================================================
__global__ __launch_bounds__(256, 1)
void looc_dist_gemm_bf16(const unsigned short* __restrict__ xb,
                         const unsigned short* __restrict__ wTb,
                         const float* __restrict__ x2, const float* __restrict__ w2,
                         float* __restrict__ out) {
    __shared__ __attribute__((aligned(16))) unsigned short As[2][BM * AST]; // 2x20KB
    __shared__ __attribute__((aligned(16))) unsigned short Bs[2][BN * BST]; // 2x10KB

    const int tid    = threadIdx.x;
    const int lane   = tid & 31;
    const int waveId = tid >> 5;
    const int waveM  = (waveId & 3) * 64;
    const int waveN  = (waveId >> 2) * 64;
    const int hf     = lane >> 4;
    const int lr     = lane & 15;

    const int blockN = blockIdx.x * BN;
    const int blockM = blockIdx.y * BM;

    // staging assignments (16B chunks): A: 1024 chunks, B: 512 chunks
    const unsigned short* ga[4]; int laOff[4];
    #pragma unroll
    for (int j = 0; j < 4; ++j) {
        int idx = tid + j * 256, row = idx >> 2, c8 = idx & 3;
        ga[j]   = xb + (size_t)(blockM + row) * IN_DIM + c8 * 8;
        laOff[j] = row * AST + c8 * 8;
    }
    const unsigned short* gb[2]; int lbOff[2];
    #pragma unroll
    for (int j = 0; j < 2; ++j) {
        int idx = tid + j * 256, n = idx >> 2, c8 = idx & 3;
        gb[j]   = wTb + (size_t)(blockN + n) * IN_DIM + c8 * 8;
        lbOff[j] = n * BST + c8 * 8;
    }

    v8f acc[4][4];
    const v8f vzero = {0.f,0.f,0.f,0.f,0.f,0.f,0.f,0.f};
    #pragma unroll
    for (int mi = 0; mi < 4; ++mi)
        #pragma unroll
        for (int ni = 0; ni < 4; ++ni) acc[mi][ni] = vzero;

    const int KT = IN_DIM / BK;

#if USE_ASYNC
    // -------- double-buffered async pipeline: one barrier per K-step ------
    #pragma unroll
    for (int j = 0; j < 4; ++j) async_ld16(ga[j], &As[0][laOff[j]]);
    #pragma unroll
    for (int j = 0; j < 2; ++j) async_ld16(gb[j], &Bs[0][lbOff[j]]);
    __builtin_amdgcn_s_wait_asynccnt(0);
    __syncthreads();

    for (int kt = 0; kt < KT; ++kt) {
        const int b = kt & 1;
        if (kt + 1 < KT) {                    // overlap next-tile DMA w/ wmma
            const int ko = (kt + 1) * BK;
            #pragma unroll
            for (int j = 0; j < 4; ++j) async_ld16(ga[j] + ko, &As[b ^ 1][laOff[j]]);
            #pragma unroll
            for (int j = 0; j < 2; ++j) async_ld16(gb[j] + ko, &Bs[b ^ 1][lbOff[j]]);
        }
#else
    // -------- synchronous fallback: two barriers, single buffer -----------
    for (int kt = 0; kt < KT; ++kt) {
        const int b = 0;
        uint4v ar[4], br[2];
        #pragma unroll
        for (int j = 0; j < 4; ++j) ar[j] = *(const uint4v*)(ga[j] + kt * BK);
        #pragma unroll
        for (int j = 0; j < 2; ++j) br[j] = *(const uint4v*)(gb[j] + kt * BK);
        __syncthreads();
        #pragma unroll
        for (int j = 0; j < 4; ++j) *(uint4v*)(&As[0][laOff[j]]) = ar[j];
        #pragma unroll
        for (int j = 0; j < 2; ++j) *(uint4v*)(&Bs[0][lbOff[j]]) = br[j];
        __syncthreads();
#endif
        // ---- fragments (ISA 7.12.2 lane layouts) and 16 WMMAs ----
        ABFrag a[4];
        #pragma unroll
        for (int mi = 0; mi < 4; ++mi) {
            const unsigned short* p = &As[b][(waveM + mi * 16 + lr) * AST + hf * 8];
            a[mi].u[0] = *(const uint4v*)(p);
            a[mi].u[1] = *(const uint4v*)(p + 16);
        }
        #pragma unroll
        for (int ni = 0; ni < 4; ++ni) {
            ABFrag bb;
            const unsigned short* p = &Bs[b][(waveN + ni * 16 + lr) * BST + hf * 8];
            bb.u[0] = *(const uint4v*)(p);
            bb.u[1] = *(const uint4v*)(p + 16);
            #pragma unroll
            for (int mi = 0; mi < 4; ++mi) {
                acc[mi][ni] = __builtin_amdgcn_wmma_f32_16x16x32_bf16(
                    false, a[mi].v, false, bb.v, (short)0, acc[mi][ni],
                    false, false);
            }
        }
#if USE_ASYNC
        __builtin_amdgcn_s_wait_asynccnt(0);
        __syncthreads();
#endif
    }

    // ---- epilogue: d = sqrt(max(|x|^2 - 2*xw + |w|^2, 0)) ----
    #pragma unroll
    for (int mi = 0; mi < 4; ++mi) {
        const int mbase = blockM + waveM + mi * 16 + hf * 8;
        float xr[8];
        #pragma unroll
        for (int r = 0; r < 8; ++r) xr[r] = x2[mbase + r];
        #pragma unroll
        for (int ni = 0; ni < 4; ++ni) {
            const int n  = blockN + waveN + ni * 16 + lr;
            const float wn = w2[n];
            #pragma unroll
            for (int r = 0; r < 8; ++r) {
                float d2 = xr[r] - 2.0f * acc[mi][ni][r] + wn;
                out[(size_t)(mbase + r) * OUT_DIM + n] = __builtin_sqrtf(fmaxf(d2, 0.0f));
            }
        }
    }
}

// ===========================================================================
// Small-workspace fallback GEMM: fp32 inputs, inline bf16 convert (round-1).
// ===========================================================================
__global__ __launch_bounds__(256, 1)
void looc_dist_gemm_f32(const float* __restrict__ x, const float* __restrict__ w,
                        const float* __restrict__ x2, const float* __restrict__ w2,
                        float* __restrict__ out) {
    __shared__ __attribute__((aligned(16))) unsigned short As[BM * AST];
    __shared__ __attribute__((aligned(16))) unsigned short Bs[BN * BST];

    const int tid    = threadIdx.x;
    const int lane   = tid & 31;
    const int waveId = tid >> 5;
    const int waveM  = (waveId & 3) * 64;
    const int waveN  = (waveId >> 2) * 64;
    const int hf     = lane >> 4;
    const int lr     = lane & 15;
    const int blockN = blockIdx.x * BN;
    const int blockM = blockIdx.y * BM;

    v8f acc[4][4];
    const v8f vzero = {0.f,0.f,0.f,0.f,0.f,0.f,0.f,0.f};
    #pragma unroll
    for (int mi = 0; mi < 4; ++mi)
        #pragma unroll
        for (int ni = 0; ni < 4; ++ni) acc[mi][ni] = vzero;

    const int KT = IN_DIM / BK;
    for (int kt = 0; kt < KT; ++kt) {
        if (kt + 1 < KT) {
            int row = tid >> 3, c4 = tid & 7;
            __builtin_prefetch(x + (size_t)(blockM + row) * IN_DIM + (kt + 1) * BK + c4 * 4, 0, 3);
        }
        __syncthreads();
        #pragma unroll
        for (int j = 0; j < 8; ++j) {
            int idx = tid + j * 256, row = idx >> 3, c4 = idx & 7;
            const float4 v = *(const float4*)(x + (size_t)(blockM + row) * IN_DIM + kt * BK + c4 * 4);
            unsigned int p0 = (unsigned int)f2bf(v.x) | ((unsigned int)f2bf(v.y) << 16);
            unsigned int p1 = (unsigned int)f2bf(v.z) | ((unsigned int)f2bf(v.w) << 16);
            *(uint2*)(&As[row * AST + c4 * 4]) = make_uint2(p0, p1);
        }
        #pragma unroll
        for (int j = 0; j < 4; ++j) {
            int idx = tid + j * 256, kr = idx >> 5, n4 = idx & 31;
            const float4 v = *(const float4*)(w + (size_t)(kt * BK + kr) * OUT_DIM + blockN + n4 * 4);
            Bs[(n4 * 4 + 0) * BST + kr] = f2bf(v.x);
            Bs[(n4 * 4 + 1) * BST + kr] = f2bf(v.y);
            Bs[(n4 * 4 + 2) * BST + kr] = f2bf(v.z);
            Bs[(n4 * 4 + 3) * BST + kr] = f2bf(v.w);
        }
        __syncthreads();

        ABFrag a[4];
        #pragma unroll
        for (int mi = 0; mi < 4; ++mi) {
            const unsigned short* p = &As[(waveM + mi * 16 + lr) * AST + hf * 8];
            a[mi].u[0] = *(const uint4v*)(p);
            a[mi].u[1] = *(const uint4v*)(p + 16);
        }
        #pragma unroll
        for (int ni = 0; ni < 4; ++ni) {
            ABFrag bb;
            const unsigned short* p = &Bs[(waveN + ni * 16 + lr) * BST + hf * 8];
            bb.u[0] = *(const uint4v*)(p);
            bb.u[1] = *(const uint4v*)(p + 16);
            #pragma unroll
            for (int mi = 0; mi < 4; ++mi)
                acc[mi][ni] = __builtin_amdgcn_wmma_f32_16x16x32_bf16(
                    false, a[mi].v, false, bb.v, (short)0, acc[mi][ni], false, false);
        }
    }

    #pragma unroll
    for (int mi = 0; mi < 4; ++mi) {
        const int mbase = blockM + waveM + mi * 16 + hf * 8;
        float xr[8];
        #pragma unroll
        for (int r = 0; r < 8; ++r) xr[r] = x2[mbase + r];
        #pragma unroll
        for (int ni = 0; ni < 4; ++ni) {
            const int n  = blockN + waveN + ni * 16 + lr;
            const float wn = w2[n];
            #pragma unroll
            for (int r = 0; r < 8; ++r) {
                float d2 = xr[r] - 2.0f * acc[mi][ni][r] + wn;
                out[(size_t)(mbase + r) * OUT_DIM + n] = __builtin_sqrtf(fmaxf(d2, 0.0f));
            }
        }
    }
}

extern "C" void kernel_launch(void* const* d_in, const int* in_sizes, int n_in,
                              void* d_out, int out_size, void* d_ws, size_t ws_size,
                              hipStream_t stream) {
    const float* x = (const float*)d_in[0];   // [8192, 4096] fp32
    const float* w = (const float*)d_in[1];   // [1, 4096, 2048] fp32
    float* out = (float*)d_out;

    float* x2 = (float*)d_ws;                                 //  8192 f
    float* w2 = x2 + B_DIM;                                   //  2048 f
    unsigned short* xb  = (unsigned short*)(w2 + OUT_DIM);    //  64 MB bf16
    unsigned short* wTb = xb + (size_t)B_DIM * IN_DIM;        //  16 MB bf16
    const size_t ws_need = (size_t)(B_DIM + OUT_DIM) * 4
                         + (size_t)B_DIM * IN_DIM * 2
                         + (size_t)OUT_DIM * IN_DIM * 2;      // ~80 MB

    dim3 grid(OUT_DIM / BN, B_DIM / BM);                      // 16 x 32

    if (ws_size >= ws_need) {
        looc_xprep<<<B_DIM / 8, 256, 0, stream>>>(x, xb, x2);
        looc_wprep<<<dim3(OUT_DIM / 64, IN_DIM / 64), 256, 0, stream>>>(w, wTb);
        looc_colnorm<<<OUT_DIM / 256, 256, 0, stream>>>(w, w2);
        looc_dist_gemm_bf16<<<grid, 256, 0, stream>>>(xb, wTb, x2, w2, out);
    } else {
        looc_rownorm<<<B_DIM / 8, 256, 0, stream>>>(x, x2);
        looc_colnorm<<<OUT_DIM / 256, 256, 0, stream>>>(w, w2);
        looc_dist_gemm_f32<<<grid, 256, 0, stream>>>(x, w, x2, w2, out);
    }
}
